// NeuralLongTermMemory_65798898975139
// MI455X (gfx1250) — matrix operand order repaired and checked by reading, not compiled, not run on hardware
//
#include <hip/hip_runtime.h>
#include <hip/hip_bf16.h>
#include <cstddef>
#include <cstdint>

typedef __attribute__((ext_vector_type(16))) __bf16 v16bf;
typedef __attribute__((ext_vector_type(8)))  float  v8f;
typedef __attribute__((ext_vector_type(4)))  float  fl4;
typedef __attribute__((ext_vector_type(4)))  unsigned ui4;
typedef __attribute__((ext_vector_type(2)))  unsigned ui2;

#define TILE 128
#define KT   32
#define LPAD 8

// ---------------- Tensor Data Mover (gfx1250 TDM) ----------------
#if __has_builtin(__builtin_amdgcn_tensor_load_to_lds)
#define HAVE_TDM 1
typedef __attribute__((ext_vector_type(4))) unsigned tdm_u4;
typedef __attribute__((ext_vector_type(8))) int      tdm_i8;
typedef __attribute__((ext_vector_type(4))) int      tdm_i4;

// DMA a (128 rows x 32 bf16) tile, row stride = lda elements, into LDS laid out
// as [128][32+8] bf16 (64B data + 16B pad per row). 2D descriptor, D# groups
// per CDNA5 ISA ch.8: data_size=2B, tile_dim0=32, tile_dim1=128,
// pad_interval=16 DWORDs (code 3), pad_amount=4 DWORDs (code 3).
__device__ __forceinline__ void tdm_load_tile_bf16(
    const unsigned short* gsrc, unsigned ldsOff, int lda)
{
  unsigned long long ga = (unsigned long long)(uintptr_t)gsrc;
  tdm_u4 g0;
  g0[0] = 1u;                                            // count=1, user mode
  g0[1] = ldsOff;                                        // lds_addr (bytes)
  g0[2] = (unsigned)(ga & 0xFFFFFFFFu);                  // global_addr[31:0]
  g0[3] = (unsigned)((ga >> 32) & 0x1FFFFFFu) | (2u << 30);  // addr[56:32] | type=2
  tdm_i8 g1;
  g1[0] = (int)((1u << 16) | (1u << 20) | (3u << 22) | (3u << 25));
          // data_size=2B | pad_enable | pad_interval=16dw | pad_amount=4dw
  g1[1] = 0;                                             // atomic barrier addr lo
  g1[2] = 0x4000;                                        // tensor_dim0 = 1<<30 (no OOB)
  g1[3] = 0x4000 | (32 << 16);                           // tensor_dim1 = 1<<30, tile_dim0=32
  g1[4] = 128;                                           // tile_dim1 = 128, tile_dim2 = 0
  g1[5] = lda;                                           // tensor_dim0_stride[31:0]
  g1[6] = 0;                                             // stride hi, dim1_stride lo
  g1[7] = 0;
  tdm_i4 gz = {0, 0, 0, 0};
#if defined(__clang_major__) && __clang_major__ >= 23
  tdm_i8 gz8 = {0, 0, 0, 0, 0, 0, 0, 0};
  __builtin_amdgcn_tensor_load_to_lds(g0, g1, gz, gz, gz8, 0);
#else
  __builtin_amdgcn_tensor_load_to_lds(g0, g1, gz, gz, 0);
#endif
}
#else
#define HAVE_TDM 0
#endif
// -----------------------------------------------------------------

__device__ __forceinline__ unsigned short f32_to_bf16_rn(float f) {
  unsigned u = __float_as_uint(f);
  u += 0x7FFFu + ((u >> 16) & 1u);          // round-to-nearest-even
  return (unsigned short)(u >> 16);
}
__device__ __forceinline__ unsigned pack_bf16x2(float a, float b) {
  unsigned ua = __float_as_uint(a); ua += 0x7FFFu + ((ua >> 16) & 1u);
  unsigned ub = __float_as_uint(b); ub += 0x7FFFu + ((ub >> 16) & 1u);
  return (ua >> 16) | (ub & 0xFFFF0000u);
}
__device__ __forceinline__ float sigmoidf_(float x) { return 1.0f / (1.0f + __expf(-x)); }
__device__ __forceinline__ float siluf_(float x)    { return x * sigmoidf_(x); }
__device__ __forceinline__ float silu_deriv_(float x) {
  float s = sigmoidf_(x); return s * (1.0f + x * (1.0f - s));
}
__device__ __forceinline__ float clip1_(float x) { return fminf(1.0f, fmaxf(-1.0f, x)); }

union Frag { v16bf v; unsigned u[8]; };

// C = opA(A) * opB(B [+ B2]) with fused epilogues. ABF/BBF: operand is bf16 (1) or f32 (0).
// EPI: 0 C0=acc; 1 C0=acc, Cb=bf16(silu); 2 Cb=bf16(silu); 3 Cb=bf16(scale*clip(acc-AUX));
//      4 Cb=bf16(acc*silu'(AUX)); 5 g=clip(acc); C0=eta*C0-lr*g; C1=(1-a)*C1+C0;
//      6 Cb=bf16(acc) with chunk-row -> sequence-row remap (CS=64, S=4096).
template<int EPI, int TA, int TB, int ABF, int BBF>
__global__ __launch_bounds__(256) void gemm_wmma_bf16(
    int M, int N, int K,
    const void* __restrict__ Ap, int lda,
    const void* __restrict__ Bp, int ldb, const float* __restrict__ B2,
    float* __restrict__ C0, float* __restrict__ C1,
    unsigned short* __restrict__ Cb, int ldc,
    const float* __restrict__ AUX,
    const float* __restrict__ ga, const float* __restrict__ gl,
    const float* __restrict__ ge, int chunk, float scaleP)
{
  (void)M; (void)N;
  __shared__ unsigned short As[TILE][KT + LPAD];   // [m][k]
  __shared__ unsigned short Bs[TILE][KT + LPAD];   // [n][k]

  const float* Af = (const float*)Ap;
  const unsigned short* Ab = (const unsigned short*)Ap;
  const float* Bf = (const float*)Bp;
  const unsigned short* Bb = (const unsigned short*)Bp;

  const int tid  = threadIdx.x;
  const int lane = tid & 31;
  const int wave = tid >> 5;          // 8 waves
  const int wr   = wave >> 1;         // 0..3 : 32-row slab
  const int wc   = wave & 1;          // 0..1 : 64-col slab
  const int half = lane >> 4;
  const int lidx = lane & 15;
  const int rowBase = blockIdx.y * TILE;
  const int colBase = blockIdx.x * TILE;
  const bool hasB2 = (B2 != nullptr);

  v8f acc[2][4];
  const v8f vzero = {0.f,0.f,0.f,0.f,0.f,0.f,0.f,0.f};
  #pragma unroll
  for (int i = 0; i < 2; ++i)
    #pragma unroll
    for (int j = 0; j < 4; ++j) acc[i][j] = vzero;

  for (int k0 = 0; k0 < K; k0 += KT) {
    const bool pf = (k0 + KT) < K;
    // ---- stage A tile (TILE rows x KT k) as bf16, layout [m][k] ----
    if constexpr (TA == 0 && ABF == 0) {
      #pragma unroll
      for (int p = 0; p < 4; ++p) {
        int idx = tid + p * 256;                 // 1024 float4 slots
        int m = idx >> 3, kq = (idx & 7) << 2;
        const float* g = Af + (size_t)(rowBase + m) * lda + (k0 + kq);
        if (p == 0 && pf) __builtin_prefetch(g + KT, 0, 1);
        fl4 a = *(const fl4*)g;
        ui2 pk; pk.x = pack_bf16x2(a.x, a.y); pk.y = pack_bf16x2(a.z, a.w);
        *(ui2*)(&As[m][kq]) = pk;
      }
    } else if constexpr (TA == 0 && ABF == 1) {
#if HAVE_TDM
      if (wave == 0) {
        tdm_load_tile_bf16(Ab + (size_t)rowBase * lda + k0,
                           (unsigned)(uintptr_t)(&As[0][0]), lda);
      }
#else
      #pragma unroll
      for (int p = 0; p < 2; ++p) {
        int idx = tid + p * 256;                 // 512 uint4 slots (8 bf16 each)
        int m = idx >> 2, kq = (idx & 3) << 3;
        const unsigned short* g = Ab + (size_t)(rowBase + m) * lda + (k0 + kq);
        if (p == 0 && pf) __builtin_prefetch(g + KT, 0, 1);
        *(ui4*)(&As[m][kq]) = *(const ui4*)g;
      }
#endif
    } else {  // TA == 1, ABF == 1 (transposed bf16)
      #pragma unroll
      for (int p = 0; p < 2; ++p) {
        int idx = tid + p * 256;
        int k = idx >> 4, mq = (idx & 15) << 3;
        const unsigned short* g = Ab + (size_t)(k0 + k) * lda + (rowBase + mq);
        if (p == 0 && pf) __builtin_prefetch(g + (size_t)KT * lda, 0, 1);
        ui4 t = *(const ui4*)g;
        #pragma unroll
        for (int e = 0; e < 4; ++e) {
          unsigned w = t[e];
          As[mq + 2 * e][k]     = (unsigned short)(w & 0xFFFFu);
          As[mq + 2 * e + 1][k] = (unsigned short)(w >> 16);
        }
      }
    }
    // ---- stage B tile (KT k x TILE cols) as bf16, layout [n][k] ----
    if constexpr (TB == 1 && BBF == 0) {
      #pragma unroll
      for (int p = 0; p < 4; ++p) {
        int idx = tid + p * 256;
        int n = idx >> 3, kq = (idx & 7) << 2;
        size_t gi = (size_t)(colBase + n) * ldb + (k0 + kq);
        if (p == 0 && pf) __builtin_prefetch(Bf + gi + KT, 0, 1);
        fl4 b = *(const fl4*)(Bf + gi);
        if (hasB2) { fl4 b2 = *(const fl4*)(B2 + gi); b += b2; }
        ui2 pk; pk.x = pack_bf16x2(b.x, b.y); pk.y = pack_bf16x2(b.z, b.w);
        *(ui2*)(&Bs[n][kq]) = pk;
      }
    } else if constexpr (TB == 0 && BBF == 0) {
      #pragma unroll
      for (int p = 0; p < 4; ++p) {
        int idx = tid + p * 256;
        int k = idx >> 5, nq = (idx & 31) << 2;
        size_t gi = (size_t)(k0 + k) * ldb + (colBase + nq);
        if (p == 0 && pf) __builtin_prefetch(Bf + gi + (size_t)KT * ldb, 0, 1);
        fl4 b = *(const fl4*)(Bf + gi);
        if (hasB2) { fl4 b2 = *(const fl4*)(B2 + gi); b += b2; }
        #pragma unroll
        for (int e = 0; e < 4; ++e) Bs[nq + e][k] = f32_to_bf16_rn(b[e]);
      }
    } else {  // TB == 0, BBF == 1
      #pragma unroll
      for (int p = 0; p < 2; ++p) {
        int idx = tid + p * 256;
        int k = idx >> 4, nq = (idx & 15) << 3;
        const unsigned short* g = Bb + (size_t)(k0 + k) * ldb + (colBase + nq);
        if (p == 0 && pf) __builtin_prefetch(g + (size_t)KT * ldb, 0, 1);
        ui4 t = *(const ui4*)g;
        #pragma unroll
        for (int e = 0; e < 4; ++e) {
          unsigned w = t[e];
          Bs[nq + 2 * e][k]     = (unsigned short)(w & 0xFFFFu);
          Bs[nq + 2 * e + 1][k] = (unsigned short)(w >> 16);
        }
      }
    }
#if HAVE_TDM
    if constexpr (TA == 0 && ABF == 1) {
      __builtin_amdgcn_s_wait_tensorcnt(0);   // TDM tile landed in LDS
    }
#endif
    __syncthreads();

    // A fragments: lane holds row m, K pairs per ISA 16-bit A 16x32 layout
    Frag af[2];
    #pragma unroll
    for (int i = 0; i < 2; ++i) {
      const unsigned* rowA = (const unsigned*)(&As[wr * 32 + i * 16 + lidx][0]);
      #pragma unroll
      for (int v = 0; v < 4; ++v) af[i].u[v]     = rowA[(half * 8 + 2 * v) >> 1];
      #pragma unroll
      for (int v = 0; v < 4; ++v) af[i].u[4 + v] = rowA[(16 + half * 8 + 2 * v) >> 1];
    }
    // B fragments: lane holds col n, K pairs per ISA 16-bit layout (same as A)
    Frag bf[4];
    #pragma unroll
    for (int j = 0; j < 4; ++j) {
      const unsigned* rowB = (const unsigned*)(&Bs[wc * 64 + j * 16 + lidx][0]);
      #pragma unroll
      for (int v = 0; v < 8; ++v) bf[j].u[v] = rowB[(v >> 2) * 8 + half * 4 + (v & 3)];
    }
    #pragma unroll
    for (int i = 0; i < 2; ++i)
      #pragma unroll
      for (int j = 0; j < 4; ++j)
        acc[i][j] = __builtin_amdgcn_wmma_f32_16x16x32_bf16(
            false, af[i].v, false, bf[j].v, (short)0, acc[i][j], false, false);
    __syncthreads();
  }

  float g_a = 0.f, g_l = 0.f, g_e = 0.f;
  if (EPI == 5) { g_a = ga[chunk]; g_l = gl[chunk]; g_e = ge[chunk]; }

  #pragma unroll
  for (int i = 0; i < 2; ++i) {
    #pragma unroll
    for (int j = 0; j < 4; ++j) {
      #pragma unroll
      for (int r = 0; r < 8; ++r) {
        int row = rowBase + wr * 32 + i * 16 + half * 8 + r;
        int col = colBase + wc * 64 + j * 16 + lidx;
        float val = acc[i][j][r];
        size_t oi = (size_t)row * ldc + col;
        if constexpr (EPI == 0) {
          C0[oi] = val;
        } else if constexpr (EPI == 1) {
          C0[oi] = val;
          Cb[oi] = f32_to_bf16_rn(siluf_(val));
        } else if constexpr (EPI == 2) {
          Cb[oi] = f32_to_bf16_rn(siluf_(val));
        } else if constexpr (EPI == 3) {
          Cb[oi] = f32_to_bf16_rn(scaleP * clip1_(val - AUX[oi]));
        } else if constexpr (EPI == 4) {
          Cb[oi] = f32_to_bf16_rn(val * silu_deriv_(AUX[oi]));
        } else if constexpr (EPI == 5) {
          float g = clip1_(val);
          float m = g_e * C0[oi] - g_l * g;
          float w = (1.0f - g_a) * C1[oi] + m;
          C0[oi] = m;
          C1[oi] = w;
        } else {  // EPI == 6: chunk-local row (b*64+c) -> token (b*4096 + n*64 + c)
          size_t so = ((size_t)(row >> 6) * 4096 + (size_t)(row & 63)) * ldc + col;
          Cb[so] = f32_to_bf16_rn(val);
        }
      }
    }
  }
}

// Per-chunk scalar gates: mean over (B, CS) of sigmoid(x . w + b), 3 gates at once.
__global__ __launch_bounds__(256) void gates_kernel(
    const float* __restrict__ x,
    const float* __restrict__ wd, const float* __restrict__ bd,
    const float* __restrict__ wl, const float* __restrict__ bl,
    const float* __restrict__ we, const float* __restrict__ be,
    float* __restrict__ alpha, float* __restrict__ lr, float* __restrict__ eta)
{
  const int n = blockIdx.x;            // chunk 0..63
  const int tid = threadIdx.x;
  __shared__ float red0[256], red1[256], red2[256];
  float s0 = 0.f, s1 = 0.f, s2 = 0.f;
  for (int t = tid; t < 512; t += 256) {     // 8 batches * 64 tokens
    int b = t >> 6, c = t & 63;
    const float* xr = x + ((size_t)b * 4096 + (size_t)n * 64 + c) * 512;
    float d0 = 0.f, d1 = 0.f, d2 = 0.f;
    for (int dd = 0; dd < 512; dd += 4) {
      fl4 xv = *(const fl4*)(xr + dd);
      fl4 w0 = *(const fl4*)(wd + dd);
      fl4 w1 = *(const fl4*)(wl + dd);
      fl4 w2 = *(const fl4*)(we + dd);
      d0 += xv.x*w0.x + xv.y*w0.y + xv.z*w0.z + xv.w*w0.w;
      d1 += xv.x*w1.x + xv.y*w1.y + xv.z*w1.z + xv.w*w1.w;
      d2 += xv.x*w2.x + xv.y*w2.y + xv.z*w2.z + xv.w*w2.w;
    }
    s0 += sigmoidf_(d0 + bd[0]);
    s1 += sigmoidf_(d1 + bl[0]);
    s2 += sigmoidf_(d2 + be[0]);
  }
  red0[tid] = s0; red1[tid] = s1; red2[tid] = s2;
  __syncthreads();
  for (int st = 128; st > 0; st >>= 1) {
    if (tid < st) { red0[tid] += red0[tid+st]; red1[tid] += red1[tid+st]; red2[tid] += red2[tid+st]; }
    __syncthreads();
  }
  if (tid == 0) {
    alpha[n] = red0[0] * (1.0f / 512.0f);
    lr[n]    = red1[0] * (1.0f / 512.0f);
    eta[n]   = red2[0] * (1.0f / 512.0f);
  }
}

// Depthwise causal conv (K=4) + bias, seq-major f32 input -> chunk-major [NC,B,CS,D].
// OBF=1: bf16 output (matrix operand); OBF=0: f32 output (epilogue operand).
template<int OBF>
__global__ __launch_bounds__(256) void conv_chunk_kernel(
    const float* __restrict__ pre, const float* __restrict__ w,
    const float* __restrict__ bias, float* __restrict__ outf,
    unsigned short* __restrict__ outb)
{
  size_t idx = (size_t)blockIdx.x * 256 + threadIdx.x;  // over 32768*512 elems
  int d = (int)(idx & 511);
  int t = (int)(idx >> 9);
  int b = t >> 12;        // / 4096
  int s = t & 4095;
  float acc = bias[d];
  #pragma unroll
  for (int j = 0; j < 4; ++j) {
    int sp = s - 3 + j;
    if (sp >= 0) acc += w[d * 4 + j] * pre[((size_t)b * 4096 + sp) * 512 + d];
  }
  int n = s >> 6, c = s & 63;
  size_t oi = ((((size_t)n * 8 + b) * 64) + c) * 512 + d;
  if constexpr (OBF) outb[oi] = f32_to_bf16_rn(acc);
  else               outf[oi] = acc;
}

extern "C" void kernel_launch(void* const* d_in, const int* in_sizes, int n_in,
                              void* d_out, int out_size, void* d_ws, size_t ws_size,
                              hipStream_t stream) {
  (void)in_sizes; (void)n_in; (void)out_size; (void)ws_size;
  const float* x    = (const float*)d_in[0];
  const float* Wk   = (const float*)d_in[1];
  const float* Wv   = (const float*)d_in[2];
  const float* Wq   = (const float*)d_in[3];
  const float* Wout = (const float*)d_in[4];
  const float* ckw  = (const float*)d_in[5];  const float* ckb = (const float*)d_in[6];
  const float* cvw  = (const float*)d_in[7];  const float* cvb = (const float*)d_in[8];
  const float* cqw  = (const float*)d_in[9];  const float* cqb = (const float*)d_in[10];
  const float* wdec = (const float*)d_in[11]; const float* bdec = (const float*)d_in[12];
  const float* wlr  = (const float*)d_in[13]; const float* blr  = (const float*)d_in[14];
  const float* weta = (const float*)d_in[15]; const float* beta = (const float*)d_in[16];
  const float* W1   = (const float*)d_in[17];
  const float* W2   = (const float*)d_in[18];
  float* out = (float*)d_out;

  const size_t TOK = 32768, D = 512, H = 1024;
  char* base = (char*)d_ws;
  size_t off = 0;
  auto alloc = [&](size_t bytes) {
    void* p = base + off; off += (bytes + 255) & ~(size_t)255; return p;
  };
  unsigned short* k_bf = (unsigned short*)alloc(TOK * D * 2);
  unsigned short* q_bf = (unsigned short*)alloc(TOK * D * 2);
  unsigned short* yseq = (unsigned short*)alloc(TOK * D * 2);
  float* v_f  = (float*)alloc(TOK * D * 4);
  float* pre  = (float*)alloc(TOK * D * 4);
  float* h1   = (float*)alloc(512 * H * 4);
  unsigned short* a1b = (unsigned short*)alloc(512 * H * 2);
  unsigned short* d2b = (unsigned short*)alloc(512 * D * 2);
  unsigned short* d1b = (unsigned short*)alloc(512 * H * 2);
  unsigned short* arb = (unsigned short*)alloc(512 * H * 2);
  float* Wd1 = (float*)alloc(H * D * 4);  float* M1 = (float*)alloc(H * D * 4);
  float* Wd2 = (float*)alloc(D * H * 4);  float* M2 = (float*)alloc(D * H * 4);
  float* ga  = (float*)alloc(64 * 4);
  float* gl  = (float*)alloc(64 * 4);
  float* ge  = (float*)alloc(64 * 4);

  // zero state each call (deterministic across graph replays)
  hipMemsetAsync(Wd1, 0, H * D * 4, stream);
  hipMemsetAsync(M1,  0, H * D * 4, stream);
  hipMemsetAsync(Wd2, 0, D * H * 4, stream);
  hipMemsetAsync(M2,  0, D * H * 4, stream);

  dim3 blk(256);
  const float* nfp = nullptr;

  // per-chunk gates
  gates_kernel<<<64, blk, 0, stream>>>(x, wdec, bdec, wlr, blr, weta, beta, ga, gl, ge);

  // projections + causal depthwise conv (chunk-major outputs; k/q bf16, v f32)
  dim3 gProj(512 / TILE, 32768 / TILE);
  const int convBlocks = (int)((TOK * D) / 256);
  gemm_wmma_bf16<0,0,1,0,0><<<gProj, blk, 0, stream>>>(32768, 512, 512, x, 512, Wk, 512, nfp,
      pre, nullptr, nullptr, 512, nfp, nfp, nfp, nfp, 0, 0.f);
  conv_chunk_kernel<1><<<convBlocks, blk, 0, stream>>>(pre, ckw, ckb, nullptr, k_bf);
  gemm_wmma_bf16<0,0,1,0,0><<<gProj, blk, 0, stream>>>(32768, 512, 512, x, 512, Wv, 512, nfp,
      pre, nullptr, nullptr, 512, nfp, nfp, nfp, nfp, 0, 0.f);
  conv_chunk_kernel<0><<<convBlocks, blk, 0, stream>>>(pre, cvw, cvb, v_f, nullptr);
  gemm_wmma_bf16<0,0,1,0,0><<<gProj, blk, 0, stream>>>(32768, 512, 512, x, 512, Wq, 512, nfp,
      pre, nullptr, nullptr, 512, nfp, nfp, nfp, nfp, 0, 0.f);
  conv_chunk_kernel<1><<<convBlocks, blk, 0, stream>>>(pre, cqw, cqb, nullptr, q_bf);

  const float scale = 2.0f / 64.0f;
  dim3 gMN(1024 / TILE, 512 / TILE);   // 512 x 1024 outputs
  dim3 gNM(512 / TILE, 1024 / TILE);   // 1024 x 512 outputs
  dim3 gSq(512 / TILE, 512 / TILE);    // 512 x 512 outputs

  for (int n = 0; n < 64; ++n) {
    const unsigned short* kc = k_bf + (size_t)n * 512 * 512;
    const unsigned short* qc = q_bf + (size_t)n * 512 * 512;
    const float*          vc = v_f  + (size_t)n * 512 * 512;
    // h1 = kc @ (W1+Wd1)^T ; a1 = silu(h1) (bf16)
    gemm_wmma_bf16<1,0,1,1,0><<<gMN, blk, 0, stream>>>(512, 1024, 512, kc, 512, W1, 512, Wd1,
        h1, nullptr, a1b, 1024, nfp, nfp, nfp, nfp, n, 0.f);
    // d2 = scale * clip(a1 @ (W2+Wd2)^T - vc) (bf16)
    gemm_wmma_bf16<3,0,1,1,0><<<gSq, blk, 0, stream>>>(512, 512, 1024, a1b, 1024, W2, 1024, Wd2,
        nullptr, nullptr, d2b, 512, vc, nfp, nfp, nfp, n, scale);
    // d1 = (d2 @ (W2+Wd2)) * silu'(h1)   [reads OLD Wd2] (bf16)
    gemm_wmma_bf16<4,0,0,1,0><<<gMN, blk, 0, stream>>>(512, 1024, 512, d2b, 512, W2, 1024, Wd2,
        nullptr, nullptr, d1b, 1024, h1, nfp, nfp, nfp, n, 0.f);
    // g1 = clip(d1^T @ kc); M1 = eta*M1 - lr*g1; Wd1 = (1-a)*Wd1 + M1
    gemm_wmma_bf16<5,1,0,1,1><<<gNM, blk, 0, stream>>>(1024, 512, 512, d1b, 1024, kc, 512, nfp,
        M1, Wd1, nullptr, 512, nfp, ga, gl, ge, n, 0.f);
    // g2 = clip(d2^T @ a1); M2 = eta*M2 - lr*g2; Wd2 = (1-a)*Wd2 + M2
    gemm_wmma_bf16<5,1,0,1,1><<<gMN, blk, 0, stream>>>(512, 1024, 512, d2b, 512, a1b, 1024, nfp,
        M2, Wd2, nullptr, 1024, nfp, ga, gl, ge, n, 0.f);
    // ar = silu(qc @ (W1+Wd1_new)^T) (bf16)
    gemm_wmma_bf16<2,0,1,1,0><<<gMN, blk, 0, stream>>>(512, 1024, 512, qc, 512, W1, 512, Wd1,
        nullptr, nullptr, arb, 1024, nfp, nfp, nfp, nfp, n, 0.f);
    // r = ar @ (W2+Wd2_new)^T -> scattered into sequence-major yseq (bf16)
    gemm_wmma_bf16<6,0,1,1,0><<<gSq, blk, 0, stream>>>(512, 512, 1024, arb, 1024, W2, 1024, Wd2,
        nullptr, nullptr, yseq + (size_t)n * 64 * 512, 512, nfp, nfp, nfp, nfp, n, 0.f);
  }

  // final: out = yseq @ Wout^T (f32 out)
  gemm_wmma_bf16<0,0,1,1,0><<<gProj, blk, 0, stream>>>(32768, 512, 512, yseq, 512, Wout, 512, nfp,
      out, nullptr, nullptr, 512, nfp, nfp, nfp, nfp, 0, 0.f);
}
